// model3_53953379172892
// MI455X (gfx1250) — matrix operand order repaired and checked by reading, not compiled, not run on hardware
//
#include <hip/hip_runtime.h>
#include <hip/hip_bf16.h>

// ---- problem constants ----
#define BB 2048
#define TT 512
#define DD 64
#define KSZ 16
#define SS 256   // KSZ*KSZ joint states

typedef __attribute__((ext_vector_type(16))) _Float16 v16h;
typedef __attribute__((ext_vector_type(8)))  float    v8f;

union V16H { v16h v; unsigned int u[8]; };

// Workgroup split-barrier with LDS-only (dscnt) drain: unlike __syncthreads(),
// this does NOT wait loadcnt, so in-flight global loads (the y_{t+1} register
// double-buffer) stay outstanding across the barrier. Mnemonics match the
// compiler's own gfx1250 barrier lowering (s_barrier_signal/wait -1).
__device__ __forceinline__ void wg_barrier_ds() {
    asm volatile("s_wait_dscnt 0x0\n\t"
                 "s_barrier_signal -1\n\t"
                 "s_barrier_wait -1" ::: "memory");
}

// ============================================================
// Prep: Wemit (hi/lo f16 split) and base = sum_d log1p(-p)
// grid = 256 states, block = 64 dims
// ============================================================
__global__ void fhmm_prep_emit(const float* __restrict__ py,
                               _Float16* __restrict__ whi,
                               _Float16* __restrict__ wlo,
                               float* __restrict__ basev) {
    const int s = blockIdx.x;
    const int d = threadIdx.x;
    float p   = py[s * DD + d];
    float lp  = __logf(p);
    float l1p = log1pf(-p);
    float w   = lp - l1p;
    _Float16 hi = (_Float16)w;
    whi[s * DD + d] = hi;
    wlo[s * DD + d] = (_Float16)(w - (float)hi);

    __shared__ float red[DD];
    red[d] = l1p;
    __syncthreads();
    #pragma unroll
    for (int o = 32; o; o >>= 1) {
        if (d < o) red[d] += red[d + o];
        __syncthreads();
    }
    if (d == 0) basev[s] = red[0];
}

// ============================================================
// Prior: Dirichlet(0.9 I + 0.1) rows on probs_w/x + Beta(0.1,0.9) on probs_y
// All gammaln terms are compile-time constants. Deterministic tree reduce.
// ============================================================
__global__ void fhmm_prior(const float* __restrict__ pw,
                           const float* __restrict__ px,
                           const float* __restrict__ py,
                           float* __restrict__ prior_out) {
    const int tid = threadIdx.x;  // 256
    float acc = 0.f;
    {   // off-diagonal conc=0.1 -> (conc-1)*log p = -0.9 log p ; diag conc=1 -> 0
        bool offdiag = ((tid >> 4) != (tid & 15));
        float a = pw[tid], b = px[tid];
        if (offdiag) acc += -0.9f * (__logf(a) + __logf(b));
    }
    for (int j = tid; j < SS * DD; j += 256) {
        float p = py[j];
        acc += -0.9f * __logf(p) - 0.1f * log1pf(-p);
    }
    __shared__ float red[256];
    red[tid] = acc;
    __syncthreads();
    #pragma unroll
    for (int o = 128; o; o >>= 1) {
        if (tid < o) red[tid] += red[tid + o];
        __syncthreads();
    }
    if (tid == 0) {
        const double g01 = 2.2527126517342059;   // lgamma(0.1)
        const double g09 = 0.0663762397347436;   // lgamma(0.9)
        const double g25 = 0.2846828704729192;   // lgamma(2.5)
        double cmat  = 16.0 * g25 - 240.0 * g01;          // per transition matrix
        double cbeta = -(g01 + g09) * (double)(SS * DD);  // Beta normalizers
        prior_out[0] = red[0] + (float)(2.0 * cmat + cbeta);
    }
}

// ============================================================
// Main scaled-forward kernel. 128 blocks x 512 threads (16 waves).
// Wave wv owns sequence row wv (recursion, alpha/tmpb are wave-private LDS)
// and N-tile wv = states [16*wv, 16*wv+16) (emission WMMA).
// Cross-wave LDS data per step: ylds (A-fragments) and logE (stage 2) only,
// so 2 split barriers per step suffice.
// ============================================================
__global__ __launch_bounds__(512)
void fhmm_forward(const int* __restrict__ seq, const int* __restrict__ lengths,
                  const float* __restrict__ probs_w, const float* __restrict__ probs_x,
                  const _Float16* __restrict__ whi, const _Float16* __restrict__ wlo,
                  const float* __restrict__ basev, float* __restrict__ ll_out) {
    __shared__ float    alpha[16][SS];   // scaled forward probs   (wave-private row)
    __shared__ float    tmpb [16][SS];   // stage-1 result [b][w'*16+x] (wave-private)
    __shared__ float    logE [16][SS];   // emission log-probs (cross-wave)
    __shared__ _Float16 ylds [16][DD];   // current observation tile (cross-wave)
    __shared__ float    Tw[KSZ][KSZ];    // Tw[w][w']
    __shared__ float    Tx[KSZ][KSZ];    // Tx[x][x']
    __shared__ int      lenS[16];

    const int tid  = threadIdx.x;
    const int b0   = blockIdx.x * 16;
    const int lane = tid & 31;
    const int wv   = tid >> 5;       // wave 0..15 == sequence row == N-tile
    const int hlf  = lane >> 4;      // lane half (WMMA layouts)
    const int ncol = lane & 15;      // WMMA M/N lane index
    const int wp   = (lane >> 1);    // 0..15: w' column owned in recursion stages
    const int xb   = (lane & 1) * 8; // x / x' sub-range base (8 wide)

    // ---------------- init ----------------
    if (tid < 256) {
        Tw[tid >> 4][tid & 15] = probs_w[tid];
        Tx[tid >> 4][tid & 15] = probs_x[tid];
    }
    if (tid < 16) lenS[tid] = lengths[b0 + tid];
    #pragma unroll
    for (int j = 0; j < 8; ++j) alpha[wv][lane * 8 + j] = 0.f;
    if (lane == 0) alpha[wv][0] = 1.0f;   // deterministic start state (0,0)
    wg_barrier_ds();

    // -------- loop-invariant state: B fragments, Tw column, length ----------
    const int sN = 16 * wv + ncol;        // this lane's output state column
    V16H bh[2], bl[2];                    // [K-chunk]
    #pragma unroll
    for (int c = 0; c < 2; ++c) {
        #pragma unroll
        for (int v = 0; v < 8; ++v) {
            int d = c * 32 + ((v & 4) ? 16 : 0) + hlf * 8 + (v & 3) * 2;   // B layout: lane=N, K-pair per VGPR (table order, matches A)
            bh[c].u[v] = *(const unsigned*)(whi + sN * DD + d);
            bl[c].u[v] = *(const unsigned*)(wlo + sN * DD + d);
        }
    }
    const float baseN = basev[sN];
    const int   mylen = lenS[wv];
    float tw[16];                          // Tw[:, wp] hoisted out of t-loop
    #pragma unroll
    for (int w = 0; w < 16; ++w) tw[w] = Tw[w][wp];

    // y staging: wave wv stages its own sequence row; int2 (2 dims) per lane
    const int d2 = lane * 2;
    const size_t yrow = (size_t)(b0 + wv) * TT * DD;
    int2 yv = *(const int2*)(seq + yrow + d2);      // t = 0, issued early

    float ll = 0.f;  // meaningful on lane 0 of each wave

    for (int t = 0; t < TT; ++t) {
        // ---- phase A: commit staged y_t -> LDS f16; issue load for t+1.
        // The load stays outstanding across the dscnt-only barrier below.
        ylds[wv][d2 + 0] = (_Float16)(float)yv.x;
        ylds[wv][d2 + 1] = (_Float16)(float)yv.y;
        int2 ynext = yv;
        if (t + 1 < TT)
            ynext = *(const int2*)(seq + yrow + (size_t)(t + 1) * DD + d2);
        wg_barrier_ds();   // ylds ready; also fences prev-iter logE reads (WAR)

        // ---- phase B1: emission logE tile via WMMA (hi+lo split) ----
        V16H a0, a1;
        #pragma unroll
        for (int v = 0; v < 8; ++v) {
            int dk = ((v & 4) ? 16 : 0) + hlf * 8 + (v & 3) * 2;  // A layout K-pair
            a0.u[v] = *(const unsigned*)&ylds[ncol][dk];
            a1.u[v] = *(const unsigned*)&ylds[ncol][32 + dk];
        }
        v8f acc;
        #pragma unroll
        for (int r = 0; r < 8; ++r) acc[r] = baseN;
        acc = __builtin_amdgcn_wmma_f32_16x16x32_f16(false, a0.v, false, bh[0].v, (short)0, acc, false, false);
        acc = __builtin_amdgcn_wmma_f32_16x16x32_f16(false, a0.v, false, bl[0].v, (short)0, acc, false, false);
        acc = __builtin_amdgcn_wmma_f32_16x16x32_f16(false, a1.v, false, bh[1].v, (short)0, acc, false, false);
        acc = __builtin_amdgcn_wmma_f32_16x16x32_f16(false, a1.v, false, bl[1].v, (short)0, acc, false, false);
        #pragma unroll
        for (int r = 0; r < 8; ++r)                 // C layout: row = r + 8*half
            logE[r + 8 * hlf][16 * wv + ncol] = acc[r];

        // ---- phase B2: stage 1 — marginalize previous w (wave-private) ----
        // tmp[b][w'*16 + x] = sum_w alpha[b][w*16 + x] * Tw[w][w']
        #pragma unroll
        for (int xo = 0; xo < 8; ++xo) {
            const int x = xb + xo;
            float s = 0.f;
            #pragma unroll
            for (int w = 0; w < 16; ++w) s += alpha[wv][w * 16 + x] * tw[w];
            tmpb[wv][wp * 16 + x] = s;
        }
        wg_barrier_ds();   // logE ready for cross-wave stage-2 reads

        // ---- phase C: stage 2 + emission combine + normalize ----
        float tr[16];
        #pragma unroll
        for (int x = 0; x < 16; ++x) tr[x] = tmpb[wv][wp * 16 + x];
        float le[8], nv[8];
        float m = -3.0e38f;
        #pragma unroll
        for (int xo = 0; xo < 8; ++xo) {
            const int xp = xb + xo;
            float s = 0.f;
            #pragma unroll
            for (int x = 0; x < 16; ++x) s += tr[x] * Tx[x][xp];
            nv[xo] = s;
            float l = logE[wv][wp * 16 + xp];
            le[xo] = l;
            m = fmaxf(m, l);
        }
        // per-sequence rowmax: full-wave32 reduction (wave == sequence)
        #pragma unroll
        for (int o = 16; o; o >>= 1) m = fmaxf(m, __shfl_xor(m, o, 32));
        float e[8], csum = 0.f;
        #pragma unroll
        for (int xo = 0; xo < 8; ++xo) {
            e[xo] = nv[xo] * __expf(le[xo] - m);
            csum += e[xo];
        }
        #pragma unroll
        for (int o = 16; o; o >>= 1) csum += __shfl_xor(csum, o, 32);

        if (t < mylen) {                  // masked update (frozen past seq end)
            float inv = 1.0f / csum;
            #pragma unroll
            for (int xo = 0; xo < 8; ++xo)
                alpha[wv][wp * 16 + xb + xo] = e[xo] * inv;
            if (lane == 0) ll += __logf(csum) + m;
        }
        // no third barrier: alpha/tmpb are wave-private; ylds/logE WAR hazards
        // are fenced by the two barriers above in the next iteration.
        yv = ynext;
    }

    // final logsumexp(alpha) == accumulated log scales (alpha normalized)
    if (lane == 0) ll_out[b0 + wv] = ll;
}

// ============================================================
// Deterministic final reduction: sum 2048 per-sequence LLs + prior
// ============================================================
__global__ void fhmm_finalize(const float* __restrict__ ll,
                              const float* __restrict__ prior_part,
                              float* __restrict__ out) {
    const int tid = threadIdx.x;  // 256
    float acc = 0.f;
    for (int j = tid; j < BB; j += 256) acc += ll[j];
    __shared__ float red[256];
    red[tid] = acc;
    __syncthreads();
    #pragma unroll
    for (int o = 128; o; o >>= 1) {
        if (tid < o) red[tid] += red[tid + o];
        __syncthreads();
    }
    if (tid == 0) out[0] = red[0] + prior_part[0];
}

extern "C" void kernel_launch(void* const* d_in, const int* in_sizes, int n_in,
                              void* d_out, int out_size, void* d_ws, size_t ws_size,
                              hipStream_t stream) {
    const int*   seq     = (const int*)d_in[0];    // (B,T,D) int32
    const int*   lengths = (const int*)d_in[1];    // (B,)    int32
    const float* pw      = (const float*)d_in[2];  // (K,K)   f32
    const float* px      = (const float*)d_in[3];  // (K,K)   f32
    const float* py      = (const float*)d_in[4];  // (K,K,D) f32
    float* out = (float*)d_out;

    char* ws = (char*)d_ws;
    _Float16* whi   = (_Float16*)(ws);                 // 256*64 f16 = 32 KB
    _Float16* wlo   = (_Float16*)(ws + 32768);         // 32 KB
    float*    basev = (float*)(ws + 65536);            // 256 f32
    float*    llb   = (float*)(ws + 66560);            // 2048 f32
    float*    prip  = (float*)(ws + 74752);            // 1 f32

    fhmm_prep_emit<<<SS, DD, 0, stream>>>(py, whi, wlo, basev);
    fhmm_prior<<<1, 256, 0, stream>>>(pw, px, py, prip);
    fhmm_forward<<<BB / 16, 512, 0, stream>>>(seq, lengths, pw, px, whi, wlo, basev, llb);
    fhmm_finalize<<<1, 256, 0, stream>>>(llb, prip, out);
}
